// HybridNN_44727789420660
// MI455X (gfx1250) — matrix operand order repaired and checked by reading, not compiled, not run on hardware
//
#include <hip/hip_runtime.h>
#include <hip/hip_bf16.h>
#include <stdint.h>

typedef __attribute__((ext_vector_type(16))) _Float16 v16h;
typedef __attribute__((ext_vector_type(8)))  _Float16 v8h;
typedef __attribute__((ext_vector_type(8)))  float    v8f;

// ---------------------------------------------------------------------------
// f32 -> f16 conversion
// ---------------------------------------------------------------------------
__global__ void cvt_f32_f16(const float* __restrict__ in, _Float16* __restrict__ out, int n) {
    int i = blockIdx.x * blockDim.x + threadIdx.x;
    if (i < n) out[i] = (_Float16)in[i];
}

// Build a 16x32 f16 A/B fragment for v_wmma_f32_16x16x32_f16 from a row-major
// row pointer (K contiguous). lane half selects which K-halves this lane owns.
__device__ __forceinline__ v16h load_frag_a(const _Float16* row, int k0, int half) {
    return *(const v16h*)(row + k0 + half * 16);
}
__device__ __forceinline__ v16h load_frag_b(const _Float16* row, int k0, int half) {
    return *(const v16h*)(row + k0 + half * 16);
}

// ---------------------------------------------------------------------------
// GEMM1: h[1024,512] = relu(x[1024,2048] @ W1^T + b1), f16 in, f16 out.
// One wave per 32x32 super-tile (2x2 register tiling of 16x16 WMMA tiles):
// each A/B fragment is reused twice -> 4 wmma per 4 fragment loads.
// 32x16 = 512 super-tiles, 4 waves/block -> 128 blocks.
// ---------------------------------------------------------------------------
__global__ __launch_bounds__(128) void gemm1_relu_wmma(
    const _Float16* __restrict__ A,   // 1024 x 2048
    const _Float16* __restrict__ Bw,  // 512 x 2048 (W1 row-major = B^T)
    const float*    __restrict__ bias,
    _Float16*       __restrict__ C)   // 1024 x 512
{
    const int Nn = 512, K = 2048;
    const int wave = threadIdx.x >> 5;
    const int lane = threadIdx.x & 31;
    const int tile = blockIdx.x * 4 + wave;     // 0..511
    const int sm = tile >> 4;                   // 0..31  (M super-tile, 32 rows)
    const int sn = tile & 15;                   // 0..15  (N super-tile, 32 cols)
    const int half = lane >> 4;
    const int l16  = lane & 15;

    const _Float16* arow0 = A  + (size_t)(sm * 32 + l16)      * K;
    const _Float16* arow1 = A  + (size_t)(sm * 32 + 16 + l16) * K;
    const _Float16* brow0 = Bw + (size_t)(sn * 32 + l16)      * K;
    const _Float16* brow1 = Bw + (size_t)(sn * 32 + 16 + l16) * K;

    v8f acc00 = {}, acc01 = {}, acc10 = {}, acc11 = {};
    for (int k0 = 0; k0 < K; k0 += 32) {
        __builtin_prefetch(arow0 + k0 + 128, 0, 1);
        __builtin_prefetch(brow0 + k0 + 128, 0, 1);
        v16h a0 = load_frag_a(arow0, k0, half);
        v16h a1 = load_frag_a(arow1, k0, half);
        v16h b0 = load_frag_b(brow0, k0, half);
        v16h b1 = load_frag_b(brow1, k0, half);
        acc00 = __builtin_amdgcn_wmma_f32_16x16x32_f16(false, a0, false, b0,
                                                       (short)0, acc00, false, false);
        acc01 = __builtin_amdgcn_wmma_f32_16x16x32_f16(false, a0, false, b1,
                                                       (short)0, acc01, false, false);
        acc10 = __builtin_amdgcn_wmma_f32_16x16x32_f16(false, a1, false, b0,
                                                       (short)0, acc10, false, false);
        acc11 = __builtin_amdgcn_wmma_f32_16x16x32_f16(false, a1, false, b1,
                                                       (short)0, acc11, false, false);
    }

    const int n0 = sn * 32 + l16;
    const int n1 = n0 + 16;
    const float bs0 = bias[n0];
    const float bs1 = bias[n1];
    #pragma unroll
    for (int r = 0; r < 8; ++r) {
        int m0 = sm * 32 + half * 8 + r;
        int m1 = m0 + 16;
        float v;
        v = acc00[r] + bs0; v = v > 0.f ? v : 0.f; C[(size_t)m0 * Nn + n0] = (_Float16)v;
        v = acc01[r] + bs1; v = v > 0.f ? v : 0.f; C[(size_t)m0 * Nn + n1] = (_Float16)v;
        v = acc10[r] + bs0; v = v > 0.f ? v : 0.f; C[(size_t)m1 * Nn + n0] = (_Float16)v;
        v = acc11[r] + bs1; v = v > 0.f ? v : 0.f; C[(size_t)m1 * Nn + n1] = (_Float16)v;
    }
}

// ---------------------------------------------------------------------------
// GEMM2: com[1024,32] = h[1024,512] @ W2^T + b2, f16 in, f32 out.
// 64x2 = 128 tiles, one wave per tile.
// ---------------------------------------------------------------------------
__global__ __launch_bounds__(128) void gemm2_wmma(
    const _Float16* __restrict__ A,   // 1024 x 512
    const _Float16* __restrict__ Bw,  // 32 x 512
    const float*    __restrict__ bias,
    float*          __restrict__ C)   // 1024 x 32
{
    const int Nn = 32, K = 512;
    const int wave = threadIdx.x >> 5;
    const int lane = threadIdx.x & 31;
    const int tile = blockIdx.x * 4 + wave;
    const int tm = tile / (Nn / 16);
    const int tn = tile % (Nn / 16);
    const int half = lane >> 4;
    const int l16  = lane & 15;

    const _Float16* arow = A  + (size_t)(tm * 16 + l16) * K;
    const _Float16* brow = Bw + (size_t)(tn * 16 + l16) * K;

    v8f acc = {};
    for (int k0 = 0; k0 < K; k0 += 32) {
        v16h av = load_frag_a(arow, k0, half);
        v16h bv = load_frag_b(brow, k0, half);
        acc = __builtin_amdgcn_wmma_f32_16x16x32_f16(false, av, false, bv,
                                                     (short)0, acc, false, false);
    }
    const int n  = tn * 16 + l16;
    const float bs = bias[n];
    #pragma unroll
    for (int r = 0; r < 8; ++r) {
        int m = tm * 16 + half * 8 + r;
        C[(size_t)m * Nn + n] = acc[r] + bs;
    }
}

// ---------------------------------------------------------------------------
// 12-qubit statevector circuit, one workgroup (256 thr) per batch row.
// State (4096 complex) lives entirely in LDS: 32 KB + 8 KB phi scratch.
// ---------------------------------------------------------------------------
__global__ __launch_bounds__(256) void circuit_kernel(
    const float* __restrict__ com,    // 1024 x 32
    const float* __restrict__ xflat,  // 1024 x 2048
    const float* __restrict__ asz,    // 2 x 12 x 3
    float*       __restrict__ out)    // 1024
{
    __shared__ float sre[4096];
    __shared__ float sim[4096];
    __shared__ float phi[2048];
    __shared__ float cA[32], sA[32];

    const int b   = blockIdx.x;
    const int tid = threadIdx.x;
    const float inv = 0.70710678118654752440f;

    // init state |0...0> + |1 0...0> (wire0 MSB)
    for (int i = tid; i < 4096; i += 256) { sre[i] = 0.f; sim[i] = 0.f; }
    __syncthreads();
    if (tid == 0) { sre[0] = inv; sre[2048] = inv; }

    // ---------------- FRQI (com, q=5) ----------------
    if (tid < 32) phi[tid] = com[b * 32 + tid];
    __syncthreads();
    #pragma unroll
    for (int s = 0; s < 5; ++s) {                 // zeta (subset-sum) transform
        if (tid < 32 && (tid & (1 << s))) phi[tid] += phi[tid - (1 << s)];
        __syncthreads();
    }
    if (tid < 32) {
        int rv = (int)(__brev((unsigned)tid) >> 27);   // 5-bit reverse
        float h = 0.5f * phi[rv];
        cA[tid] = __cosf(h); sA[tid] = __sinf(h);
    }
    __syncthreads();
    // state view (2, 32, 64): i = a*64 + r pairs with i+2048
    for (int j = 0; j < 8; ++j) {
        int i = tid + 256 * j;                    // 0..2047
        int a = i >> 6;
        float c = cA[a], sn = sA[a];
        float s0r = sre[i], s0i = sim[i];
        float s1r = sre[i + 2048], s1i = sim[i + 2048];
        sre[i]        = c * s0r - sn * s1r;  sim[i]        = c * s0i - sn * s1i;
        sre[i + 2048] = sn * s0r + c * s1r;  sim[i + 2048] = sn * s0i + c * s1i;
    }
    __syncthreads();

    // ---------------- strongly entangling, 2 layers ----------------
    for (int l = 0; l < 2; ++l) {
        for (int w = 0; w < 12; ++w) {
            const float* p = asz + (l * 12 + w) * 3;
            float ph = p[0], th = p[1], om = p[2];
            float ct = __cosf(0.5f * th), st = __sinf(0.5f * th);
            float apo = 0.5f * (ph + om), amo = 0.5f * (ph - om);
            float c1 = __cosf(apo), s1v = __sinf(apo);
            float c2 = __cosf(amo), s2v = __sinf(amo);
            float m00r =  c1 * ct, m00i = -s1v * ct;
            float m01r = -c2 * st, m01i = -s2v * st;
            float m10r =  c2 * st, m10i = -s2v * st;
            float m11r =  c1 * ct, m11i =  s1v * ct;
            const int Rb = 11 - w;
            const int Rm = (1 << Rb) - 1;
            for (int j = 0; j < 8; ++j) {
                int pdx = tid + 256 * j;          // pair 0..2047
                int r   = pdx & Rm;
                int li  = pdx >> Rb;
                int n0  = (li << (Rb + 1)) + r;
                int n1  = n0 + (1 << Rb);
                float a0r = sre[n0], a0i = sim[n0];
                float a1r = sre[n1], a1i = sim[n1];
                sre[n0] = m00r * a0r - m00i * a0i + m01r * a1r - m01i * a1i;
                sim[n0] = m00r * a0i + m00i * a0r + m01r * a1i + m01i * a1r;
                sre[n1] = m10r * a0r - m10i * a0i + m11r * a1r - m11i * a1i;
                sim[n1] = m10r * a0i + m10i * a0r + m11r * a1i + m11i * a1r;
            }
            __syncthreads();
        }
        // CNOT ring permutation: out[n] = state[perm(n)]
        const int rr = (l % 11) + 1;
        float tr[16], ti[16];
        for (int j = 0; j < 16; ++j) {
            int n = tid + 256 * j;
            int y = n;
            #pragma unroll
            for (int i2 = 11; i2 >= 0; --i2) {
                int cm = 1 << (11 - i2);
                int tm = 1 << (11 - ((i2 + rr) % 12));
                if (y & cm) y ^= tm;
            }
            tr[j] = sre[y]; ti[j] = sim[y];
        }
        __syncthreads();
        for (int j = 0; j < 16; ++j) {
            int n = tid + 256 * j;
            sre[n] = tr[j]; sim[n] = ti[j];
        }
        __syncthreads();
    }

    // ---------------- FRQI (image, q=11) ----------------
    for (int j = 0; j < 8; ++j) {
        int i = tid + 256 * j;
        phi[i] = xflat[(size_t)b * 2048 + i];
    }
    __syncthreads();
    for (int s = 0; s < 11; ++s) {                // zeta transform, 2048 elems
        int bit = 1 << s;
        for (int j = 0; j < 8; ++j) {
            int i = tid + 256 * j;
            if (i & bit) phi[i] += phi[i - bit];
        }
        __syncthreads();
    }
    for (int j = 0; j < 8; ++j) {
        int a  = tid + 256 * j;                   // 0..2047, R=1
        int rv = (int)(__brev((unsigned)a) >> 21);// 11-bit reverse
        float h = 0.5f * phi[rv];
        float c = __cosf(h), sn = __sinf(h);
        float s0r = sre[a], s0i = sim[a];
        float s1r = sre[a + 2048], s1i = sim[a + 2048];
        sre[a]        = c * s0r - sn * s1r;  sim[a]        = c * s0i - sn * s1i;
        sre[a + 2048] = sn * s0r + c * s1r;  sim[a + 2048] = sn * s0i + c * s1i;
    }
    __syncthreads();

    // ---------------- signed probability reduction ----------------
    float acc = 0.f;
    for (int j = 0; j < 8; ++j) {
        int i = tid + 256 * j;
        acc -= sre[i] * sre[i] + sim[i] * sim[i];
        int k = i + 2048;
        acc += sre[k] * sre[k] + sim[k] * sim[k];
    }
    phi[tid] = acc;
    __syncthreads();
    for (int s = 128; s > 0; s >>= 1) {
        if (tid < s) phi[tid] += phi[tid + s];
        __syncthreads();
    }
    if (tid == 0) out[b] = phi[0];
}

// ---------------------------------------------------------------------------
// launch
// ---------------------------------------------------------------------------
extern "C" void kernel_launch(void* const* d_in, const int* in_sizes, int n_in,
                              void* d_out, int out_size, void* d_ws, size_t ws_size,
                              hipStream_t stream) {
    (void)in_sizes; (void)n_in; (void)out_size; (void)ws_size;
    const float* x   = (const float*)d_in[0];   // 1024 x 2048
    const float* W1  = (const float*)d_in[1];   // 512 x 2048
    const float* b1  = (const float*)d_in[2];   // 512
    const float* W2  = (const float*)d_in[3];   // 32 x 512
    const float* b2  = (const float*)d_in[4];   // 32
    const float* asz = (const float*)d_in[5];   // 2 x 12 x 3
    float* out = (float*)d_out;                 // 1024

    // workspace carve-out (f16 elements)
    _Float16* xh  = (_Float16*)d_ws;                       // 2,097,152
    _Float16* w1h = xh  + 2097152;                         // 1,048,576
    _Float16* w2h = w1h + 1048576;                         //    16,384
    _Float16* hh  = w2h + 16384;                           //   524,288
    float*    com = (float*)(hh + 524288);                 //    32,768 f32

    cvt_f32_f16<<<(2097152 + 255) / 256, 256, 0, stream>>>(x,  xh,  2097152);
    cvt_f32_f16<<<(1048576 + 255) / 256, 256, 0, stream>>>(W1, w1h, 1048576);
    cvt_f32_f16<<<(16384   + 255) / 256, 256, 0, stream>>>(W2, w2h, 16384);

    // GEMM1: 32x16 = 512 super-tiles (32x32 each), 4 waves/block -> 128 blocks
    gemm1_relu_wmma<<<128, 128, 0, stream>>>(xh, w1h, b1, hh);
    // GEMM2: 64x2 = 128 tiles -> 32 blocks
    gemm2_wmma<<<32, 128, 0, stream>>>(hh, w2h, b2, com);

    circuit_kernel<<<1024, 256, 0, stream>>>(com, x, asz, out);
}
